// PathConvStack_21131239097225
// MI455X (gfx1250) — matrix operand order repaired and checked by reading, not compiled, not run on hardware
//
#include <hip/hip_runtime.h>
#include <hip/hip_bf16.h>

#define NUM_UIJK  500000
#define NUM_IJKL  1000000
#define NUM_IJK   250000
#define NUM_UIJKL 2000000
#define HN   8      // heads
#define PC   128    // pair channels
#define PIL  8      // MAX_PI_LENGTH
#define FD   72     // HN*(PIL+1)
#define FDP  96     // FD padded to multiple of 32 (bf16 WMMA K)
#define LSTR 104    // LDS K-stride (pad for bank spread, 16B-align preserved)

typedef __attribute__((ext_vector_type(2)))  float  v2f;
typedef __attribute__((ext_vector_type(8)))  float  v8f;
typedef __attribute__((ext_vector_type(16))) __bf16 v16bf;

// ---------------------------------------------------------------- utilities
__global__ void k_fill(float* __restrict__ p, long n, float v) {
    long i = (long)blockIdx.x * blockDim.x + threadIdx.x;
    if (i < n) p[i] = v;
}

// ------------------------------------------------- [N,128] @ [128,8] via WMMA f32
// One wave = one 16-row tile. W (128x8) zero-padded to 128x16 in LDS.
__global__ __launch_bounds__(64)
void k_gemm_c128h8(const float* __restrict__ X, const float* __restrict__ W,
                   float* __restrict__ Y) {
    __shared__ float wk[PC][16];
    int tid = threadIdx.x;
    for (int i = tid; i < PC * 16; i += 64) {
        int k = i >> 4, n = i & 15;
        wk[k][n] = (n < HN) ? W[k * HN + n] : 0.0f;
    }
    __syncthreads();

    int wave = tid >> 5, lane = tid & 31;
    int row0 = (blockIdx.x * 2 + wave) * 16;
    int m = lane & 15, h = lane >> 4;     // A: row m, K-half h ; B: col m, K-half h
    const float* xr = X + (size_t)(row0 + m) * PC;

    v8f c = {0.f, 0.f, 0.f, 0.f, 0.f, 0.f, 0.f, 0.f};
#pragma unroll 8
    for (int kk = 0; kk < PC; kk += 4) {
        v2f a, b;
        a.x = xr[kk + 2 * h];
        a.y = xr[kk + 2 * h + 1];
        b.x = wk[kk + 2 * h][m];
        b.y = wk[kk + 2 * h + 1][m];
        c = __builtin_amdgcn_wmma_f32_16x16x4_f32(false, a, false, b,
                                                  (short)0, c, false, false);
    }
    if (m < HN) {
#pragma unroll
        for (int i = 0; i < 8; ++i)
            Y[(size_t)(row0 + 8 * h + i) * HN + m] = c[i];
    }
}

// --------------------------------------------- segment max via float-bits atomics
__global__ void k_segmax(const float* __restrict__ logits, const int* __restrict__ seg,
                         float* __restrict__ segmax) {
    long i = (long)blockIdx.x * blockDim.x + threadIdx.x;
    if (i >= (long)NUM_IJKL * HN) return;
    int row = (int)(i >> 3), hh = (int)(i & 7);
    float v = logits[i];
    float* addr = segmax + (size_t)seg[row] * HN + hh;
    if (v >= 0.0f) atomicMax((int*)addr, __float_as_int(v));
    else           atomicMin((unsigned int*)addr, (unsigned int)__float_as_int(v));
}

// --------------------------------------------- e = exp(x - max); sum via atomics
__global__ void k_expsum(float* __restrict__ e, const int* __restrict__ seg,
                         const float* __restrict__ segmax, float* __restrict__ segsum) {
    long i = (long)blockIdx.x * blockDim.x + threadIdx.x;
    if (i >= (long)NUM_IJKL * HN) return;
    int row = (int)(i >> 3), hh = (int)(i & 7);
    size_t s = (size_t)seg[row] * HN + hh;
    float ev = __expf(e[i] - segmax[s]);
    e[i] = ev;
    atomicAdd(&segsum[s], ev);
}

// ---------------------- kernelU[u] = e[idxU_ijkl[u]] / segsum[seg[idxU_ijkl[u]]]
__global__ void k_makekernel(const float* __restrict__ e, const float* __restrict__ segsum,
                             const int* __restrict__ idxU_ijkl, const int* __restrict__ seg,
                             float* __restrict__ kernelU) {
    int u = blockIdx.x * blockDim.x + threadIdx.x;
    if (u >= NUM_UIJKL) return;
    int ij = idxU_ijkl[u];
    int s  = seg[ij];
    const float4* ep = (const float4*)(e + (size_t)ij * HN);
    const float4* sp = (const float4*)(segsum + (size_t)s * HN);
    float4 e0 = ep[0], e1 = ep[1], s0 = sp[0], s1 = sp[1];
    float4 r0 = {e0.x / s0.x, e0.y / s0.y, e0.z / s0.z, e0.w / s0.w};
    float4 r1 = {e1.x / s1.x, e1.y / s1.y, e1.z / s1.z, e1.w / s1.w};
    float4* kp = (float4*)(kernelU + (size_t)u * HN);
    kp[0] = r0; kp[1] = r1;
}

// ---------------------- out_next[dst] += kernelU[u] * out_prev[src] (L2 atomics)
__global__ void k_pathconv(const float* __restrict__ kernelU, const float* __restrict__ out_prev,
                           float* __restrict__ out_next, const int* __restrict__ idx_src,
                           const int* __restrict__ idx_dst) {
    int u = blockIdx.x * blockDim.x + threadIdx.x;
    if (u >= NUM_UIJKL) return;
    const float4* kp = (const float4*)(kernelU + (size_t)u * HN);
    int s = idx_src[u], d = idx_dst[u];
    const float4* gp = (const float4*)(out_prev + (size_t)s * HN);
    float4 k0 = kp[0], k1 = kp[1], g0 = gp[0], g1 = gp[1];
    float* dp = out_next + (size_t)d * HN;
    atomicAdd(dp + 0, k0.x * g0.x);
    atomicAdd(dp + 1, k0.y * g0.y);
    atomicAdd(dp + 2, k0.z * g0.z);
    atomicAdd(dp + 3, k0.w * g0.w);
    atomicAdd(dp + 4, k1.x * g1.x);
    atomicAdd(dp + 5, k1.y * g1.y);
    atomicAdd(dp + 6, k1.z * g1.z);
    atomicAdd(dp + 7, k1.w * g1.w);
}

// ------------------------------------------------ MLP: bf16 WMMA, f32 accumulate
// X[r,c] = outbuf[c/8][r][c%8] (concat of 9 layers). One wave = one 16-row tile.
__global__ __launch_bounds__(64)
void k_mlp(const float* __restrict__ outbuf, const float* __restrict__ W1,
           const float* __restrict__ b1, const float* __restrict__ W2,
           const float* __restrict__ b2, const float* __restrict__ prop,
           float* __restrict__ dout) {
    __shared__ __bf16 w1t[80][LSTR];      // W1^T: [out col 0..79][K 0..95], zero padded
    __shared__ __bf16 w2t[PC][LSTR];      // W2^T: [out col 0..127][K 0..95]
    __shared__ __bf16 hst[2][16][LSTR];   // per-wave GELU(H) stage, A-frag friendly

    int tid = threadIdx.x;
    for (int i = tid; i < 80 * LSTR; i += 64) {
        int c = i / LSTR, k = i % LSTR;
        w1t[c][k] = (c < FD && k < FD) ? (__bf16)W1[k * FD + c] : (__bf16)0.0f;
    }
    for (int i = tid; i < PC * LSTR; i += 64) {
        int c = i / LSTR, k = i % LSTR;
        w2t[c][k] = (k < FD) ? (__bf16)W2[k * PC + c] : (__bf16)0.0f;
    }
    for (int i = tid; i < 2 * 16 * LSTR; i += 64) {
        int k = i % LSTR;
        if (k >= FD) ((__bf16*)hst)[i] = (__bf16)0.0f;   // zero K-pad region
    }
    __syncthreads();

    int wave = tid >> 5, lane = tid & 31;
    int row0 = (blockIdx.x * 2 + wave) * 16;
    int m = lane & 15, h = lane >> 4;
    size_t r = (size_t)(row0 + m);

    // ---------------- GEMM1: H = X(16x72) @ W1(72x72), K padded to 96
    v8f c1[5];
#pragma unroll
    for (int t = 0; t < 5; ++t) c1[t] = (v8f){0.f,0.f,0.f,0.f,0.f,0.f,0.f,0.f};

#pragma unroll
    for (int ks = 0; ks < 3; ++ks) {
        int kk = ks * 32;
        // A fragment: e<8 -> layer L0 heads 0..7 ; e>=8 -> layer L1 heads 0..7
        int L0 = kk / 8 + h;
        int L1 = L0 + 2;
        union { float4 f[2]; float s[8]; } x0, x1;
        x0.f[0] = x0.f[1] = x1.f[0] = x1.f[1] = (float4){0.f, 0.f, 0.f, 0.f};
        if (L0 <= PIL) {
            const float4* p = (const float4*)(outbuf + ((size_t)L0 * NUM_UIJK + r) * HN);
            x0.f[0] = p[0]; x0.f[1] = p[1];
        }
        if (L1 <= PIL) {
            const float4* p = (const float4*)(outbuf + ((size_t)L1 * NUM_UIJK + r) * HN);
            x1.f[0] = p[0]; x1.f[1] = p[1];
        }
        v16bf a;
#pragma unroll
        for (int e = 0; e < 8; ++e) a[e] = (__bf16)x0.s[e];
#pragma unroll
        for (int e = 0; e < 8; ++e) a[8 + e] = (__bf16)x1.s[e];

#pragma unroll
        for (int t = 0; t < 5; ++t) {
            const __bf16* bp = &w1t[16 * t + m][kk + 8 * h];
            union { v16bf v; float4 f[2]; } b;
            b.f[0] = *(const float4*)bp;
            b.f[1] = *(const float4*)(bp + 16);
            c1[t] = __builtin_amdgcn_wmma_f32_16x16x32_bf16(false, a, false, b.v,
                                                            (short)0, c1[t], false, false);
        }
    }

    // bias + exact GELU -> stage H as bf16 in A-fragment layout
#pragma unroll
    for (int t = 0; t < 5; ++t) {
        int col = 16 * t + m;
        if (col < FD) {
            float bias = b1[col];
#pragma unroll
            for (int i = 0; i < 8; ++i) {
                float x = c1[t][i] + bias;
                float g = 0.5f * x * (1.0f + erff(x * 0.7071067811865475f));
                hst[wave][8 * h + i][col] = (__bf16)g;
            }
        }
    }
    __syncthreads();

    // ---------------- GEMM2: Y = H(16x72) @ W2(72x128) + b2 + prop
    v8f c2[8];
#pragma unroll
    for (int t = 0; t < 8; ++t) c2[t] = (v8f){0.f,0.f,0.f,0.f,0.f,0.f,0.f,0.f};

#pragma unroll
    for (int ks = 0; ks < 3; ++ks) {
        int kk = ks * 32;
        const __bf16* hp = &hst[wave][m][0];
        union { v16bf v; float4 f[2]; } a;
        a.f[0] = *(const float4*)(hp + kk + 8 * h);        // e=0..7 : K = kk+8h+e
        a.f[1] = *(const float4*)(hp + kk + 16 + 8 * h);   // e=8..15: K = kk+16+8h+(e-8)
#pragma unroll
        for (int t = 0; t < 8; ++t) {
            const __bf16* bp = &w2t[16 * t + m][kk + 8 * h];
            union { v16bf v; float4 f[2]; } b;
            b.f[0] = *(const float4*)bp;
            b.f[1] = *(const float4*)(bp + 16);
            c2[t] = __builtin_amdgcn_wmma_f32_16x16x32_bf16(false, a.v, false, b.v,
                                                            (short)0, c2[t], false, false);
        }
    }

#pragma unroll
    for (int t = 0; t < 8; ++t) {
        int col = 16 * t + m;
        float bias = b2[col];
#pragma unroll
        for (int i = 0; i < 8; ++i) {
            size_t rr = (size_t)(row0 + 8 * h + i);
            dout[rr * PC + col] = c2[t][i] + bias + prop[rr * PC + col];
        }
    }
}

// ------------------------------------------------------------------- launcher
extern "C" void kernel_launch(void* const* d_in, const int* in_sizes, int n_in,
                              void* d_out, int out_size, void* d_ws, size_t ws_size,
                              hipStream_t stream) {
    const float* prop_attr   = (const float*)d_in[0];
    const float* stereo_attr = (const float*)d_in[1];
    const float* W_value     = (const float*)d_in[2];
    const float* W_kernel    = (const float*)d_in[3];
    const float* mlp_W1      = (const float*)d_in[4];
    const float* mlp_b1      = (const float*)d_in[5];
    const float* mlp_W2      = (const float*)d_in[6];
    const float* mlp_b2      = (const float*)d_in[7];
    const int* idx_ijkl_jkl  = (const int*)d_in[8];
    const int* idxU_ijkl     = (const int*)d_in[9];
    const int* idxU_Uijk     = (const int*)d_in[10];
    const int* idxU_ujkl     = (const int*)d_in[11];
    float* out = (float*)d_out;

    char* ws = (char*)d_ws;
    float* logits  = (float*)(ws);                        // 1e6*8   = 32,000,000 B (reused for e)
    float* segmax  = (float*)(ws + 32000000);             // 250k*8  =  8,000,000 B
    float* segsum  = (float*)(ws + 40000000);             // 250k*8  =  8,000,000 B
    float* kernelU = (float*)(ws + 48000000);             // 2e6*8   = 64,000,000 B
    float* outbuf  = (float*)(ws + 112000000);            // 9*500k*8= 144,000,000 B

    // init: segmax = -inf, segsum = 0, out layers 1..8 = 0
    {
        long n = (long)NUM_IJK * HN;
        k_fill<<<(int)((n + 255) / 256), 256, 0, stream>>>(segmax, n, -INFINITY);
        k_fill<<<(int)((n + 255) / 256), 256, 0, stream>>>(segsum, n, 0.0f);
        long nz = (long)PIL * NUM_UIJK * HN;
        k_fill<<<(int)((nz + 255) / 256), 256, 0, stream>>>(outbuf + (size_t)NUM_UIJK * HN, nz, 0.0f);
    }

    // logits = stereo_attr @ W_kernel ; out[0] = prop_attr @ W_value
    k_gemm_c128h8<<<NUM_IJKL / 32, 64, 0, stream>>>(stereo_attr, W_kernel, logits);
    k_gemm_c128h8<<<NUM_UIJK / 32, 64, 0, stream>>>(prop_attr, W_value, outbuf);

    // segment softmax (max -> exp/sum -> gather-normalize)
    {
        long n = (long)NUM_IJKL * HN;
        k_segmax<<<(int)((n + 255) / 256), 256, 0, stream>>>(logits, idx_ijkl_jkl, segmax);
        k_expsum<<<(int)((n + 255) / 256), 256, 0, stream>>>(logits, idx_ijkl_jkl, segmax, segsum);
    }
    k_makekernel<<<(NUM_UIJKL + 255) / 256, 256, 0, stream>>>(logits, segsum, idxU_ijkl,
                                                              idx_ijkl_jkl, kernelU);

    // 8 path-convolution iterations (segment_sum via L2-resident f32 atomics)
    for (int l = 0; l < PIL; ++l) {
        const float* prev = outbuf + (size_t)l * NUM_UIJK * HN;
        float* next       = outbuf + (size_t)(l + 1) * NUM_UIJK * HN;
        k_pathconv<<<(NUM_UIJKL + 255) / 256, 256, 0, stream>>>(kernelU, prev, next,
                                                                idxU_Uijk, idxU_ujkl);
    }

    // GELU-MLP + residual
    k_mlp<<<NUM_UIJK / 32, 64, 0, stream>>>(outbuf, mlp_W1, mlp_b1, mlp_W2, mlp_b2,
                                            prop_attr, out);
}
